// VectorQuantizer_55353538511074
// MI455X (gfx1250) — matrix operand order repaired and checked by reading, not compiled, not run on hardware
//
#include <hip/hip_runtime.h>
#include <hip/hip_bf16.h>

typedef __attribute__((ext_vector_type(16))) _Float16 v16h;
typedef __attribute__((ext_vector_type(8)))  float    v8f;

#define NB    16
#define CH    192
#define HW    4096          // 64*64
#define NTOT  65536         // NB*HW
#define KCODE 1024
#define CT    6             // 192/32 c-tiles
#define JT    64            // 1024/16 j-tiles
#define ROWS_PER_BLK 128
#define EPSN  1e-8f

union HalfPack { _Float16 h[2]; unsigned u; };

// ---------------------------------------------------------------------------
// Prep 1: pack codebook [1024,192] f32 -> f16 WMMA B-fragments in d_ws.
// B layout (16-bit 32x16): cq=co>>3; lane = N + ((cq&1)<<4);
// slot = (co&7) + ((cq>>1)<<3)  (same k order as A).
// Flat: bp[(((jtile*6 + ctile)*32 + lane)*16 + slot]
// ---------------------------------------------------------------------------
__global__ __launch_bounds__(256) void vq_pack_codebook(const float* __restrict__ cb,
                                                        _Float16* __restrict__ bp) {
  int tid = blockIdx.x * 256 + threadIdx.x;          // over 1024*96 c-pairs
  if (tid >= KCODE * (CH / 2)) return;
  int j  = tid / (CH / 2);
  int c  = (tid % (CH / 2)) * 2;
  float a0 = cb[j * CH + c];
  float a1 = cb[j * CH + c + 1];
  int jtile = j >> 4, jm = j & 15;
  int ctile = c >> 5, co = c & 31;
  int cq   = co >> 3;
  int lane = jm + ((cq & 1) << 4);
  int slot = (co & 7) + ((cq >> 1) << 3);            // even (c even)
  HalfPack pk; pk.h[0] = (_Float16)a0; pk.h[1] = (_Float16)a1;
  unsigned* bpu = (unsigned*)bp;
  int off = ((jtile * CT + ctile) * 32 + lane) * 16 + slot; // in halfs
  bpu[off >> 1] = pk.u;
}

// ---------------------------------------------------------------------------
// Prep 2: c2[j] = ||codebook[j]||^2
// ---------------------------------------------------------------------------
__global__ __launch_bounds__(256) void vq_c2(const float* __restrict__ cb,
                                             float* __restrict__ c2) {
  int j = blockIdx.x * 256 + threadIdx.x;
  if (j >= KCODE) return;
  float s = 0.f;
  const float* row = cb + (size_t)j * CH;
  for (int c = 0; c < CH; ++c) { float v = row[c]; s += v * v; }
  c2[j] = s;
}

// ---------------------------------------------------------------------------
// Main: per block 128 rows, 8 waves, each wave one 16-row tile vs all 1024
// codes via v_wmma_f32_16x16x32_f16 (double-buffered B w/ pinned schedule,
// dual accumulators); argmin + gather writeback.
// ---------------------------------------------------------------------------
__global__ __launch_bounds__(256)
__attribute__((amdgpu_waves_per_eu(2, 4)))
void vq_main(const float* __restrict__ latent,
             const float* __restrict__ cb,
             const _Float16* __restrict__ bp,
             const float* __restrict__ c2,
             float* __restrict__ outq,
             float* __restrict__ outi) {
  // A fragments for 8 waves * 6 ctiles * 32 lanes * 16 halfs = 48 KB
  __shared__ _Float16 afrag[8 * CT * 32 * 16];
  __shared__ float    norm2[ROWS_PER_BLK];
  __shared__ int      bestj_s[ROWS_PER_BLK];

  const int t  = threadIdx.x;
  const int n0 = blockIdx.x * ROWS_PER_BLK;
  const int bI = n0 >> 12;                 // batch index (n0 / 4096)
  const int hw0 = n0 & (HW - 1);
  const float* xbase = latent + (size_t)bI * CH * HW + hw0;

  if (t < ROWS_PER_BLK) norm2[t] = 0.f;
  __syncthreads();

  // ---- cooperative load: 128 rows x 96 c-pairs, f32 -> f16 A-fragments ----
  {
    unsigned* afu = (unsigned*)afrag;
    const int r  = t & 127;               // fixed row per thread (coalesced)
    const int w  = r >> 4, m = r & 15;
    float nsum = 0.f;
#pragma unroll 4
    for (int i = 0; i < 48; ++i) {
      int cp = (t >> 7) + 2 * i;          // c-pair index
      int c  = cp * 2;
      float a0 = xbase[(size_t)c * HW + r];
      float a1 = xbase[(size_t)(c + 1) * HW + r];
      nsum += a0 * a0 + a1 * a1;
      // A layout (16-bit 16x32, per ISA): cq=co>>3;
      // lane = m + ((cq&1)<<4); slot = (co&7) + ((cq>>1)<<3)
      int ctile = c >> 5, co = c & 31;
      int cq   = co >> 3;
      int lane = m + ((cq & 1) << 4);
      int slot = (co & 7) + ((cq >> 1) << 3);   // even
      HalfPack pk; pk.h[0] = (_Float16)a0; pk.h[1] = (_Float16)a1;
      int off = ((w * CT + ctile) * 32 + lane) * 16 + slot;
      afu[off >> 1] = pk.u;
    }
    atomicAdd(&norm2[r], nsum);           // two contributors per row
  }
  __syncthreads();

  // ---- per-wave WMMA GEMM + running argmin ----
  const int wave = t >> 5;
  const int lane = t & 31;
  const int jl   = lane & 15;
  const int rhalf = (lane >> 4) << 3;     // +8 rows for hi half-wave

  v16h a[CT];
  {
    const v16h* ap = (const v16h*)&afrag[wave * CT * 32 * 16];
#pragma unroll
    for (int ct = 0; ct < CT; ++ct) a[ct] = ap[ct * 32 + lane];
  }

  float inv2[8];
#pragma unroll
  for (int v = 0; v < 8; ++v) {
    float nn = norm2[wave * 16 + rhalf + v];
    inv2[v] = 2.0f / (__builtin_sqrtf(nn) + EPSN);  // folds the -2x factor
  }

  float bestv[8];
  int   besti[8];
#pragma unroll
  for (int v = 0; v < 8; ++v) { bestv[v] = 3.0e38f; besti[v] = 0; }

  // B fragment base for this lane (v16h units)
  const v16h* bbase = (const v16h*)bp;

  v16h b0[CT], b1[CT];
  float c2_0, c2_1;

  // prime buffer 0 with j-tile 0
#pragma unroll
  for (int ct = 0; ct < CT; ++ct) b0[ct] = bbase[ct * 32 + lane];
  c2_0 = c2[jl];
  __builtin_amdgcn_sched_barrier(0);

  for (int jt = 0; jt < JT; jt += 2) {
    // ---- prefetch j-tile jt+1 into buffer 1 ----
    {
      const v16h* bn = bbase + (size_t)(jt + 1) * CT * 32;
#pragma unroll
      for (int ct = 0; ct < CT; ++ct) b1[ct] = bn[ct * 32 + lane];
      c2_1 = c2[(jt + 1) * 16 + jl];
    }
    __builtin_amdgcn_sched_barrier(0);   // keep these loads above the WMMAs
    // ---- compute j-tile jt with buffer 0 (dual accumulators) ----
    {
      v8f acc0 = {}, acc1 = {};
#pragma unroll
      for (int ct = 0; ct < CT; ct += 2) {
        acc0 = __builtin_amdgcn_wmma_f32_16x16x32_f16(
            false, a[ct], false, b0[ct], (short)0, acc0, false, false);
        acc1 = __builtin_amdgcn_wmma_f32_16x16x32_f16(
            false, a[ct + 1], false, b0[ct + 1], (short)0, acc1, false, false);
      }
      int j = jt * 16 + jl;
#pragma unroll
      for (int v = 0; v < 8; ++v) {
        float sc = c2_0 - (acc0[v] + acc1[v]) * inv2[v];
        if (sc < bestv[v]) { bestv[v] = sc; besti[v] = j; }
      }
    }
    __builtin_amdgcn_sched_barrier(0);
    // ---- prefetch j-tile jt+2 into buffer 0 (clamped on last iter) ----
    {
      int jn = (jt + 2 < JT) ? (jt + 2) : 0;   // branchless clamp, no divergence
      const v16h* bn = bbase + (size_t)jn * CT * 32;
#pragma unroll
      for (int ct = 0; ct < CT; ++ct) b0[ct] = bn[ct * 32 + lane];
      c2_0 = c2[jn * 16 + jl];
    }
    __builtin_amdgcn_sched_barrier(0);   // keep these loads above the WMMAs
    // ---- compute j-tile jt+1 with buffer 1 ----
    {
      v8f acc0 = {}, acc1 = {};
#pragma unroll
      for (int ct = 0; ct < CT; ct += 2) {
        acc0 = __builtin_amdgcn_wmma_f32_16x16x32_f16(
            false, a[ct], false, b1[ct], (short)0, acc0, false, false);
        acc1 = __builtin_amdgcn_wmma_f32_16x16x32_f16(
            false, a[ct + 1], false, b1[ct + 1], (short)0, acc1, false, false);
      }
      int j = (jt + 1) * 16 + jl;
#pragma unroll
      for (int v = 0; v < 8; ++v) {
        float sc = c2_1 - (acc0[v] + acc1[v]) * inv2[v];
        if (sc < bestv[v]) { bestv[v] = sc; besti[v] = j; }
      }
    }
    __builtin_amdgcn_sched_barrier(0);
  }

  // ---- 16-lane min-reduce (columns of the tile); halves stay separate ----
#pragma unroll
  for (int msk = 8; msk >= 1; msk >>= 1) {
#pragma unroll
    for (int v = 0; v < 8; ++v) {
      float ov = __shfl_xor(bestv[v], msk, 32);
      int   oi = __shfl_xor(besti[v], msk, 32);
      if (ov < bestv[v] || (ov == bestv[v] && oi < besti[v])) {
        bestv[v] = ov; besti[v] = oi;
      }
    }
  }
  if (jl == 0) {
#pragma unroll
    for (int v = 0; v < 8; ++v) bestj_s[wave * 16 + rhalf + v] = besti[v];
  }
  __syncthreads();

  // ---- outputs: indices (as float) + gathered codebook rows (strided) ----
  if (t < ROWS_PER_BLK) outi[n0 + t] = (float)bestj_s[t];

  {
    float* obase = outq + (size_t)bI * CH * HW + hw0;
    const int r = t & 127;
    const float* crow = cb + (size_t)bestj_s[r] * CH;
#pragma unroll 4
    for (int i = 0; i < 48; ++i) {
      int c = ((t >> 7) + 2 * i) * 2;
      obase[(size_t)c * HW + r]       = crow[c];
      obase[(size_t)(c + 1) * HW + r] = crow[c + 1];
    }
  }
}

// ---------------------------------------------------------------------------
extern "C" void kernel_launch(void* const* d_in, const int* in_sizes, int n_in,
                              void* d_out, int out_size, void* d_ws, size_t ws_size,
                              hipStream_t stream) {
  const float* latent = (const float*)d_in[0];   // [16,192,64,64] f32
  const float* cb     = (const float*)d_in[1];   // [1024,192] f32

  _Float16* bp = (_Float16*)d_ws;                         // 384 KB packed B
  float*    c2 = (float*)((char*)d_ws + (size_t)KCODE * CH * 2); // 4 KB

  float* outq = (float*)d_out;                            // [16,192,64,64]
  float* outi = outq + (size_t)NB * CH * HW;              // [16,64,64] as float

  vq_pack_codebook<<<(KCODE * (CH / 2) + 255) / 256, 256, 0, stream>>>(cb, bp);
  vq_c2<<<(KCODE + 255) / 256, 256, 0, stream>>>(cb, c2);
  vq_main<<<NTOT / ROWS_PER_BLK, 256, 0, stream>>>(latent, cb, bp, c2, outq, outi);
}
